// GatedAttentionUnit_66322884985391
// MI455X (gfx1250) — compile-verified, matrix operand order checked by reading, not run on hardware
//
#include <hip/hip_runtime.h>
#include <hip/hip_bf16.h>
#include <math.h>

// ---------------------------------------------------------------------------
// GAU block for gfx1250 (MI455X).  All GEMM-shaped work runs through
// v_wmma_f32_16x16x32_bf16 (bf16 inputs, fp32 accumulate).  Main GEMMs use
// double-buffered GLOBAL_LOAD_ASYNC_TO_LDS_B128 staging (ASYNCcnt-tracked,
// CDNA5 ISA) so WMMA overlaps the next K-slab's copy.
// ---------------------------------------------------------------------------

typedef __bf16 bf16;
typedef __attribute__((ext_vector_type(8)))  __bf16 v8bf;
typedef __attribute__((ext_vector_type(16))) __bf16 v16bf;
typedef __attribute__((ext_vector_type(8)))  float  v8f;

#define DMODEL 1024
#define DEXP   2048
#define DHEAD  128
#define BATCH  4
#define SEQ    2048
#define NROWS  (BATCH*SEQ)   // 8192

// A/B fragment load (16-bit, 16x32):
// lane = 16*h + m ; element j (0..15) -> K = 16*(j/8) + 8*h + (j%8)
// => two contiguous 16B reads at cols {8h, 16+8h}.  ld in elements.
__device__ __forceinline__ v16bf frag_ld(const bf16* __restrict__ base, int ld) {
  const int lane = threadIdx.x & 31;
  const int m = lane & 15, h = lane >> 4;
  const bf16* p = base + m * ld + 8 * h;
  v8bf lo = *(const v8bf*)p;
  v8bf hi = *(const v8bf*)(p + 16);
  return __builtin_shufflevector(lo, hi, 0,1,2,3,4,5,6,7,8,9,10,11,12,13,14,15);
}

__device__ __forceinline__ v8f wmma_bf16(v16bf a, v16bf b, v8f c) {
  return __builtin_amdgcn_wmma_f32_16x16x32_bf16(
      /*neg_a=*/false, a, /*neg_b=*/false, b,
      /*c_mod=*/(short)0, c, /*reuse_a=*/false, /*reuse_b=*/false);
}

// Async 16B global -> LDS copy (no VGPR round-trip; tracked by ASYNCcnt).
// VDST carries the per-lane LDS byte offset (low 32 bits of the flat LDS
// address are the within-allocation offset on AMDGCN).
__device__ __forceinline__ void async_ld16(void* lds, const void* gptr) {
  unsigned l = (unsigned)(unsigned long long)lds;
  unsigned long long g = (unsigned long long)gptr;
  asm volatile("global_load_async_to_lds_b128 %0, %1, off"
               :: "v"(l), "v"(g) : "memory");
}
__device__ __forceinline__ void wait_async0() {
  asm volatile("s_wait_asynccnt 0x0" ::: "memory");
}

// ---------------------------------------------------------------------------
// Weight prep: fp32 [rows][cols] -> bf16 transposed [cols][rows]
// ---------------------------------------------------------------------------
__global__ void transpose_to_bf16_kernel(const float* __restrict__ src,
                                         bf16* __restrict__ dst,
                                         int rows, int cols) {
  int idx = blockIdx.x * 256 + threadIdx.x;
  if (idx >= rows * cols) return;
  int r = idx / cols, c = idx - r * cols;
  dst[(size_t)c * rows + r] = (bf16)src[idx];
}

// ---------------------------------------------------------------------------
// LayerNorm: fp32 [NROWS][DMODEL] -> bf16
// ---------------------------------------------------------------------------
__global__ __launch_bounds__(256)
void layernorm_kernel(const float* __restrict__ x,
                      const float* __restrict__ gamma,
                      const float* __restrict__ beta,
                      bf16* __restrict__ xn) {
  const int row = blockIdx.x;
  const int tid = threadIdx.x;
  const float* xr = x + (size_t)row * DMODEL;
  float s = 0.f, ss = 0.f;
  for (int i = tid; i < DMODEL; i += 256) { float v = xr[i]; s += v; ss += v * v; }
  __shared__ float rs[256], rss[256];
  rs[tid] = s; rss[tid] = ss;
  __syncthreads();
  for (int o = 128; o > 0; o >>= 1) {
    if (tid < o) { rs[tid] += rs[tid + o]; rss[tid] += rss[tid + o]; }
    __syncthreads();
  }
  const float mu  = rs[0] * (1.f / DMODEL);
  const float var = rss[0] * (1.f / DMODEL) - mu * mu;
  const float inv = rsqrtf(var + 1e-5f);
  for (int i = tid; i < DMODEL; i += 256) {
    float v = (xr[i] - mu) * inv * gamma[i] + beta[i];
    xn[(size_t)row * DMODEL + i] = (bf16)v;
  }
}

// ---------------------------------------------------------------------------
// Generic WMMA GEMM.  A [M][K] bf16 row-major, Bt [N][K] bf16 row-major
// (i.e. B column-contiguous).  Wave tile 32x64 (2x4 C fragments), waves
// arranged (BM/32) x (BN/64).  LDS rows padded to 40 elems (80B) so the
// lane-striped 16B fragment reads hit disjoint bank groups.  K-slabs of 32
// are double-buffered via async global->LDS copies.
// ---------------------------------------------------------------------------
enum { EPI_SILU = 0, EPI_NONE = 1, EPI_RESID = 2 };

template<int BM, int BN, int EPI>
__global__ __launch_bounds__(32*(BM/32)*(BN/64))
void gemm_bf16_kernel(const bf16* __restrict__ A, const bf16* __restrict__ Bt,
                      const int M, const int N, const int K,
                      bf16* __restrict__ Dbf, float* __restrict__ Df,
                      const float* __restrict__ Res) {
  constexpr int WM = BM / 32, WN = BN / 64, NT = 32 * WM * WN;
  constexpr int LD = 40;
  __shared__ __align__(16) bf16 sA[2][BM * LD];
  __shared__ __align__(16) bf16 sB[2][BN * LD];
  const int tid  = threadIdx.x;
  const int lane = tid & 31, w = tid >> 5;
  const int wm = w % WM, wn = w / WM;
  const long bm0 = (long)blockIdx.y * BM;
  const long bn0 = (long)blockIdx.x * BN;

  const v8f vzero = {0.f,0.f,0.f,0.f,0.f,0.f,0.f,0.f};
  v8f acc[2][4];
  #pragma unroll
  for (int mi = 0; mi < 2; ++mi)
    #pragma unroll
    for (int ni = 0; ni < 4; ++ni) acc[mi][ni] = vzero;

  // Issue one K-slab (32 cols) of async copies into buffer `buf`.
  auto stage = [&](int buf, int k0) {
    #pragma unroll 1
    for (int i = tid; i < BM * 4; i += NT) {
      int r = i >> 2, c = (i & 3) << 3;
      async_ld16(&sA[buf][r * LD + c], &A[(bm0 + r) * (long)K + k0 + c]);
    }
    #pragma unroll 1
    for (int i = tid; i < BN * 4; i += NT) {
      int r = i >> 2, c = (i & 3) << 3;
      async_ld16(&sB[buf][r * LD + c], &Bt[(bn0 + r) * (long)K + k0 + c]);
    }
  };

  stage(0, 0);
  wait_async0();
  __syncthreads();

  int cur = 0;
  for (int k0 = 0; k0 < K; k0 += 32) {
    if (k0 + 32 < K) stage(cur ^ 1, k0 + 32);   // prefetch next slab

    v16bf af[2], bfr[4];
    #pragma unroll
    for (int mi = 0; mi < 2; ++mi) af[mi]  = frag_ld(&sA[cur][(wm*32 + mi*16) * LD], LD);
    #pragma unroll
    for (int ni = 0; ni < 4; ++ni) bfr[ni] = frag_ld(&sB[cur][(wn*64 + ni*16) * LD], LD);
    #pragma unroll
    for (int mi = 0; mi < 2; ++mi)
      #pragma unroll
      for (int ni = 0; ni < 4; ++ni)
        acc[mi][ni] = wmma_bf16(af[mi], bfr[ni], acc[mi][ni]);

    wait_async0();       // my slab-copies landed in LDS
    __syncthreads();     // everyone done reading cur / writing cur^1
    cur ^= 1;
  }

  const int half = lane >> 4, nn = lane & 15;
  #pragma unroll
  for (int mi = 0; mi < 2; ++mi) {
    #pragma unroll
    for (int ni = 0; ni < 4; ++ni) {
      #pragma unroll
      for (int r = 0; r < 8; ++r) {
        long row = bm0 + wm*32 + mi*16 + r + 8*half;
        long col = bn0 + wn*64 + ni*16 + nn;
        float xv = acc[mi][ni][r];
        if (EPI == EPI_SILU) {
          xv = xv / (1.f + __expf(-xv));
          Dbf[row * N + col] = (bf16)xv;
        } else if (EPI == EPI_NONE) {
          Dbf[row * N + col] = (bf16)xv;
        } else {
          Df[row * N + col] = xv + Res[row * N + col];
        }
      }
    }
  }
}

// ---------------------------------------------------------------------------
// RoPE in place on bf16 [NROWS][DHEAD]; half = 64.
// ---------------------------------------------------------------------------
__global__ __launch_bounds__(64)
void rope_kernel(bf16* __restrict__ t) {
  const int row = blockIdx.x;
  const int i   = threadIdx.x;            // 0..63
  const int pos = row & (SEQ - 1);
  const float inv = __powf(10000.f, -((float)(2 * i) / (float)DHEAD));
  const float ang = (float)pos * inv;
  const float c = __cosf(ang), s = __sinf(ang);
  bf16* p = t + (size_t)row * DHEAD;
  const float x1 = (float)p[i], x2 = (float)p[i + 64];
  p[i]      = (bf16)(x1 * c - x2 * s);
  p[i + 64] = (bf16)(x1 * s + x2 * c);
}

// ---------------------------------------------------------------------------
// Fused causal relu^2 attention + gate.
//   out[b, t, e] = u[b,t,e] * sum_{t'<=t} relu(q_t . k_t' / sqrt(s))^2 * v[b,t',e]
// Block: 64 q-rows x 256 e-cols, 8 waves.  Key blocks of 32 streamed; scores
// computed by WMMA (one 16x16 tile per wave), relu^2'd, staged to LDS as
// bf16 A-fragments; v block transposed into LDS as the B operand.
// ---------------------------------------------------------------------------
__global__ __launch_bounds__(256)
void attn_gate_kernel(const bf16* __restrict__ q, const bf16* __restrict__ k,
                      const bf16* __restrict__ v, const bf16* __restrict__ u,
                      bf16* __restrict__ gated) {
  const int b  = blockIdx.z;
  const int q0 = blockIdx.y * 64;
  const int e0 = blockIdx.x * 256;
  const bf16* qp = q + (size_t)b * SEQ * DHEAD;
  const bf16* kp = k + (size_t)b * SEQ * DHEAD;
  const bf16* vp = v + (size_t)b * SEQ * DEXP;
  const bf16* up = u + (size_t)b * SEQ * DEXP;
  bf16*       gp = gated + (size_t)b * SEQ * DEXP;

  constexpr int LD = 40;
  __shared__ __align__(16) bf16 sS[64 * LD];    // relu^2 scores, bf16
  __shared__ __align__(16) bf16 sV[256 * LD];   // v block transposed [e][key]

  const int tid = threadIdx.x;
  const int lane = tid & 31, w = tid >> 5;
  const int wm = w & 1, wn = w >> 1;            // output tiling: 2 x 4 waves
  const int sm = w & 3, sn = w >> 2;            // score tiling:  4 x 2 waves
  const int half = lane >> 4, nn = lane & 15;

  const v8f vzero = {0.f,0.f,0.f,0.f,0.f,0.f,0.f,0.f};
  v8f acc[2][4];
  #pragma unroll
  for (int mi = 0; mi < 2; ++mi)
    #pragma unroll
    for (int ni = 0; ni < 4; ++ni) acc[mi][ni] = vzero;

  // q fragments for this wave's score tile (rows q0+16*sm, K=0..127)
  v16bf qf[4];
  #pragma unroll
  for (int ks = 0; ks < 4; ++ks)
    qf[ks] = frag_ld(qp + (size_t)(q0 + sm*16) * DHEAD + ks*32, DHEAD);

  const int kblk_max = (q0 + 63) >> 5;          // causal upper key block
  for (int kblk = 0; kblk <= kblk_max; ++kblk) {
    const int kbase = kblk * 32;
    __syncthreads();                            // LDS reuse fence

    // stage v block transposed: sV[e][kk] = v[kbase+kk][e0+e]
    #pragma unroll 1
    for (int i = tid; i < 32 * 256; i += 256) {
      int kk = i >> 8, e = i & 255;
      sV[e * LD + kk] = vp[(size_t)(kbase + kk) * DEXP + e0 + e];
    }

    // scores: one 16x16 tile per wave over [64q x 32key]
    v8f sc = vzero;
    #pragma unroll
    for (int ks = 0; ks < 4; ++ks) {
      v16bf kf = frag_ld(kp + (size_t)(kbase + sn*16) * DHEAD + ks*32, DHEAD);
      sc = wmma_bf16(qf[ks], kf, sc);
    }
    const int rloc = sm*16 + 8*half;
    const int cloc = sn*16 + nn;
    const int key  = kbase + cloc;
    #pragma unroll
    for (int r = 0; r < 8; ++r) {
      float xv = sc[r] * 0.08838834764831845f;  // 1/sqrt(128)
      xv = (key <= q0 + rloc + r && xv > 0.f) ? xv * xv : 0.f;
      sS[(rloc + r) * LD + cloc] = (bf16)xv;
    }
    __syncthreads();

    // accumulate: attn_tile[32x32-of-64] @ vT
    v16bf af[2], bfr[4];
    #pragma unroll
    for (int mi = 0; mi < 2; ++mi) af[mi]  = frag_ld(&sS[(wm*32 + mi*16) * LD], LD);
    #pragma unroll
    for (int ni = 0; ni < 4; ++ni) bfr[ni] = frag_ld(&sV[(wn*64 + ni*16) * LD], LD);
    #pragma unroll
    for (int mi = 0; mi < 2; ++mi)
      #pragma unroll
      for (int ni = 0; ni < 4; ++ni)
        acc[mi][ni] = wmma_bf16(af[mi], bfr[ni], acc[mi][ni]);
  }

  // epilogue: gate by u and store bf16
  #pragma unroll
  for (int mi = 0; mi < 2; ++mi) {
    #pragma unroll
    for (int ni = 0; ni < 4; ++ni) {
      #pragma unroll
      for (int r = 0; r < 8; ++r) {
        int row = q0 + wm*32 + mi*16 + r + 8*half;
        int col = e0 + wn*64 + ni*16 + nn;
        float g = acc[mi][ni][r] * (float)up[(size_t)row * DEXP + col];
        gp[(size_t)row * DEXP + col] = (bf16)g;
      }
    }
  }
}

// ---------------------------------------------------------------------------
extern "C" void kernel_launch(void* const* d_in, const int* in_sizes, int n_in,
                              void* d_out, int out_size, void* d_ws, size_t ws_size,
                              hipStream_t stream) {
  (void)in_sizes; (void)n_in; (void)out_size; (void)ws_size;
  const float* x     = (const float*)d_in[0];
  const float* W_U   = (const float*)d_in[1];
  const float* W_V   = (const float*)d_in[2];
  const float* W_Q   = (const float*)d_in[3];
  const float* W_K   = (const float*)d_in[4];
  const float* W_O   = (const float*)d_in[5];
  const float* gamma = (const float*)d_in[6];
  const float* beta  = (const float*)d_in[7];
  // d_in[8] (mask) unused: causal mask applied analytically.
  float* out = (float*)d_out;

  char* ws = (char*)d_ws;
  size_t off = 0;
  auto alloc = [&](size_t bytes) -> void* {
    void* p = ws + off;
    off = (off + bytes + 255) & ~(size_t)255;
    return p;
  };
  bf16* xn    = (bf16*)alloc((size_t)NROWS * DMODEL * 2);
  bf16* ubuf  = (bf16*)alloc((size_t)NROWS * DEXP * 2);
  bf16* vbuf  = (bf16*)alloc((size_t)NROWS * DEXP * 2);
  bf16* qbuf  = (bf16*)alloc((size_t)NROWS * DHEAD * 2);
  bf16* kbuf  = (bf16*)alloc((size_t)NROWS * DHEAD * 2);
  bf16* gbuf  = (bf16*)alloc((size_t)NROWS * DEXP * 2);
  bf16* WUt   = (bf16*)alloc((size_t)DEXP * DMODEL * 2);
  bf16* WVt   = (bf16*)alloc((size_t)DEXP * DMODEL * 2);
  bf16* WQt   = (bf16*)alloc((size_t)DHEAD * DEXP * 2);
  bf16* WKt   = (bf16*)alloc((size_t)DHEAD * DEXP * 2);
  bf16* WOt   = (bf16*)alloc((size_t)DMODEL * DEXP * 2);

  // 1) weight prep (bf16, transposed so B operand is K-contiguous)
  transpose_to_bf16_kernel<<<(DMODEL*DEXP + 255)/256, 256, 0, stream>>>(W_U, WUt, DMODEL, DEXP);
  transpose_to_bf16_kernel<<<(DMODEL*DEXP + 255)/256, 256, 0, stream>>>(W_V, WVt, DMODEL, DEXP);
  transpose_to_bf16_kernel<<<(DEXP*DHEAD + 255)/256, 256, 0, stream>>>(W_Q, WQt, DEXP, DHEAD);
  transpose_to_bf16_kernel<<<(DEXP*DHEAD + 255)/256, 256, 0, stream>>>(W_K, WKt, DEXP, DHEAD);
  transpose_to_bf16_kernel<<<(DEXP*DMODEL + 255)/256, 256, 0, stream>>>(W_O, WOt, DEXP, DMODEL);

  // 2) LayerNorm -> bf16
  layernorm_kernel<<<NROWS, 256, 0, stream>>>(x, gamma, beta, xn);

  // 3) u = silu(xn @ W_U), v = silu(xn @ W_V)   [8192 x 2048, K=1024]
  gemm_bf16_kernel<64,256,EPI_SILU><<<dim3(DEXP/256, NROWS/64), 256, 0, stream>>>(
      xn, WUt, NROWS, DEXP, DMODEL, ubuf, nullptr, nullptr);
  gemm_bf16_kernel<64,256,EPI_SILU><<<dim3(DEXP/256, NROWS/64), 256, 0, stream>>>(
      xn, WVt, NROWS, DEXP, DMODEL, vbuf, nullptr, nullptr);

  // 4) q = u @ W_Q, k = v @ W_K   [8192 x 128, K=2048]
  gemm_bf16_kernel<64,128,EPI_NONE><<<dim3(DHEAD/128, NROWS/64), 128, 0, stream>>>(
      ubuf, WQt, NROWS, DHEAD, DEXP, qbuf, nullptr, nullptr);
  gemm_bf16_kernel<64,128,EPI_NONE><<<dim3(DHEAD/128, NROWS/64), 128, 0, stream>>>(
      vbuf, WKt, NROWS, DHEAD, DEXP, kbuf, nullptr, nullptr);

  // 5) RoPE in place
  rope_kernel<<<NROWS, 64, 0, stream>>>(qbuf);
  rope_kernel<<<NROWS, 64, 0, stream>>>(kbuf);

  // 6) fused causal relu^2 attention + gate -> gated bf16
  attn_gate_kernel<<<dim3(DEXP/256, SEQ/64, BATCH), 256, 0, stream>>>(
      qbuf, kbuf, vbuf, ubuf, gbuf);

  // 7) out = gated @ W_O + residual   [8192 x 1024, K=2048], fp32 output
  gemm_bf16_kernel<64,256,EPI_RESID><<<dim3(DMODEL/256, NROWS/64), 256, 0, stream>>>(
      gbuf, WOt, NROWS, DMODEL, DEXP, nullptr, out, x);
}
